// CheriBlock_55422257988105
// MI455X (gfx1250) — matrix operand order repaired and checked by reading, not compiled, not run on hardware
//
#include <hip/hip_runtime.h>
#include <math.h>

// ---------------- problem constants (reference: N=8, L=8192, C=512) --------
#define NB    8
#define LL    8192
#define CCH   512
#define CCH2  1024
#define MROWS (NB * LL)          // 65536 flattened rows

typedef __attribute__((ext_vector_type(16))) __bf16        v16bf;
typedef __attribute__((ext_vector_type(8)))  float         v8f;
typedef __attribute__((ext_vector_type(4)))  unsigned int  u32x4;

union AFrag { u32x4 q[2]; v16bf v; };
union Pack4 { __bf16 h[4]; unsigned long long u; };

// ---------------- kernel 1: zero stats + convert weights to bf16 -----------
__global__ __launch_bounds__(256) void prep_kernel(const float* __restrict__ W1,
                                                   const float* __restrict__ W2,
                                                   __bf16* __restrict__ w1b,
                                                   __bf16* __restrict__ w2b,
                                                   float* __restrict__ stats) {
    int idx = blockIdx.x * 256 + threadIdx.x;
    if (blockIdx.x == 0 && threadIdx.x < 16) stats[threadIdx.x] = 0.0f;
    if (idx < CCH * CCH2) {
        w1b[idx] = (__bf16)W1[idx];
        w2b[idx] = (__bf16)W2[idx];
    }
}

// ---------------- kernel 2: conv recompute + per-sample sum / sumsq --------
__global__ __launch_bounds__(256) void conv_stats_kernel(const float* __restrict__ X,
                                                         const float* __restrict__ cw,
                                                         const int* __restrict__ dil,
                                                         float* __restrict__ stats) {
    const int d = dil[0];
    const long base4 = (long)blockIdx.x * 2048;      // float4 index; 2^20 float4 per n
    const float4* px = (const float4*)X;
    const float4* pw = (const float4*)cw;            // rows: w0 | w1 | w2, each 128 float4
    float s = 0.0f, sq = 0.0f;
    for (int i = 0; i < 8; ++i) {
        long e4 = base4 + threadIdx.x + i * 256;
        int  c4 = (int)(e4 & 127);
        long g  = e4 >> 7;                           // flattened row
        int  l  = (int)(g & (LL - 1));
        float4 xc = px[g * 128 + c4];
        float4 xm = make_float4(0.f, 0.f, 0.f, 0.f);
        float4 xp = make_float4(0.f, 0.f, 0.f, 0.f);
        if (l >= d)       xm = px[(g - d) * 128 + c4];
        if (l + d < LL)   xp = px[(g + d) * 128 + c4];
        float4 w0 = pw[c4], w1 = pw[128 + c4], w2 = pw[256 + c4];
        float cx = xm.x * w0.x + xc.x * w1.x + xp.x * w2.x;
        float cy = xm.y * w0.y + xc.y * w1.y + xp.y * w2.y;
        float cz = xm.z * w0.z + xc.z * w1.z + xp.z * w2.z;
        float cww = xm.w * w0.w + xc.w * w1.w + xp.w * w2.w;
        s  += cx + cy + cz + cww;
        sq += cx * cx + cy * cy + cz * cz + cww * cww;
    }
    __shared__ float rs[256], rq[256];
    rs[threadIdx.x] = s; rq[threadIdx.x] = sq;
    __syncthreads();
    for (int st = 128; st > 0; st >>= 1) {
        if (threadIdx.x < st) { rs[threadIdx.x] += rs[threadIdx.x + st];
                                rq[threadIdx.x] += rq[threadIdx.x + st]; }
        __syncthreads();
    }
    if (threadIdx.x == 0) {
        int n = (int)(base4 >> 20);                  // 2^20 float4 per sample
        atomicAdd(&stats[2 * n + 0], rs[0]);
        atomicAdd(&stats[2 * n + 1], rq[0]);
    }
}

// ---------------- kernel 3: mean / rstd per sample --------------------------
__global__ void finalize_kernel(const float* __restrict__ stats, float* __restrict__ ms) {
    int n = threadIdx.x;
    if (n < NB) {
        const float inv = 1.0f / (float)(LL * CCH);
        float mean = stats[2 * n] * inv;
        float var  = stats[2 * n + 1] * inv - mean * mean;
        ms[2 * n + 0] = mean;
        ms[2 * n + 1] = rsqrtf(var + 1e-3f);
    }
}

// ---------------- kernel 4: fused norm + MLP (bf16 WMMA) + residual ---------
#define BM        64
#define SA_STRIDE 520           // 512 + 8 pad -> 1040 B rows, bank-conflict-free b128
#define SH_STRIDE 1032          // 1024 + 8 pad -> 2064 B rows
#define LDS_BYTES ((BM * SA_STRIDE + BM * SH_STRIDE) * 2)   // 198656 B < 320 KB WGP LDS

__global__ __launch_bounds__(256) void mlp_kernel(const float* __restrict__ X,
                                                  const float* __restrict__ cw,
                                                  const __bf16* __restrict__ w1b,
                                                  const __bf16* __restrict__ w2b,
                                                  const float* __restrict__ gamma,
                                                  const float* __restrict__ ms,
                                                  const int* __restrict__ dil,
                                                  float* __restrict__ out) {
    extern __shared__ __bf16 smem[];
    __bf16* sA = smem;                         // [BM][SA_STRIDE] normalized conv, bf16
    __bf16* sH = smem + BM * SA_STRIDE;        // [BM][SH_STRIDE] gelu(h), bf16

    const int t    = threadIdx.x;
    const int lane = t & 31;
    const int w    = t >> 5;                   // wave id 0..7
    const int row0 = blockIdx.x * BM;          // 64 rows, all inside one sample (64 | 8192)
    const int n    = row0 >> 13;
    const float mean = ms[2 * n], rstd = ms[2 * n + 1];
    const int d = dil[0];

    // ---- stage: recompute conv, normalize, quantize to bf16 into sA -------
    {
        const float4* px = (const float4*)X;
        const float4* pw = (const float4*)cw;
        for (int i = 0; i < (BM * CCH / 4) / 256; ++i) {   // 32 iterations
            int  e4  = t + i * 256;
            int  c4  = e4 & 127;
            int  row = e4 >> 7;
            long g   = row0 + row;
            int  l   = (int)(g & (LL - 1));
            float4 xc = px[g * 128 + c4];
            float4 xm = make_float4(0.f, 0.f, 0.f, 0.f);
            float4 xp = make_float4(0.f, 0.f, 0.f, 0.f);
            if (l >= d)     xm = px[(g - d) * 128 + c4];
            if (l + d < LL) xp = px[(g + d) * 128 + c4];
            float4 w0 = pw[c4], w1 = pw[128 + c4], w2 = pw[256 + c4];
            Pack4 p;
            p.h[0] = (__bf16)(((xm.x * w0.x + xc.x * w1.x + xp.x * w2.x) - mean) * rstd);
            p.h[1] = (__bf16)(((xm.y * w0.y + xc.y * w1.y + xp.y * w2.y) - mean) * rstd);
            p.h[2] = (__bf16)(((xm.z * w0.z + xc.z * w1.z + xp.z * w2.z) - mean) * rstd);
            p.h[3] = (__bf16)(((xm.w * w0.w + xc.w * w1.w + xp.w * w2.w) - mean) * rstd);
            *(unsigned long long*)&sA[row * SA_STRIDE + c4 * 4] = p.u;
        }
    }
    __syncthreads();

    const int cb    = lane & 15;       // B-frag column / A-frag row / D-frag N
    const int khalf = lane >> 4;       // 0: lanes 0-15, 1: lanes 16-31

    // ---- GEMM1: h = sA(64x512) x W1^T (512x1024), gelu -> sH --------------
    for (int jt = w; jt < 64; jt += 8) {
        int col0 = jt * 16;
        v8f acc[4] = {};
        for (int kc = 0; kc < 16; ++kc) {
            int k0 = kc * 32;
            // B 32x16: lane holds column cb, 16 consecutive K at k0 + khalf*16
            v16bf bv = *(const v16bf*)(w1b + (size_t)(col0 + cb) * CCH + k0 + khalf * 16);
#pragma unroll
            for (int mt = 0; mt < 4; ++mt) {
                // A 16x32: lane holds row cb; 16 consecutive K at k0 + khalf*16
                AFrag af;
                const __bf16* pa = &sA[(mt * 16 + cb) * SA_STRIDE + k0 + khalf * 16];
                af.q[0] = *(const u32x4*)pa;
                af.q[1] = *(const u32x4*)(pa + 8);
                acc[mt] = __builtin_amdgcn_wmma_f32_16x16x32_bf16(
                    false, af.v, false, bv, (short)0, acc[mt], false, false);
            }
        }
#pragma unroll
        for (int mt = 0; mt < 4; ++mt) {
            int mrow = mt * 16 + khalf * 8;     // D: VGPR r -> M = r + 8*khalf
#pragma unroll
            for (int r = 0; r < 8; ++r) {
                float x = acc[mt][r];
                float gl = 0.5f * x * (1.0f + tanhf(0.7978845608f * (x + 0.044715f * x * x * x)));
                sH[(mrow + r) * SH_STRIDE + col0 + cb] = (__bf16)gl;
            }
        }
    }
    __syncthreads();

    // ---- GEMM2: x_mlp = sH(64x1024) x W2^T (1024x512) + residual ----------
    for (int ct = w; ct < 32; ct += 8) {
        int c0 = ct * 16;
        v8f acc[4] = {};
        for (int kc = 0; kc < 32; ++kc) {
            int k0 = kc * 32;
            v16bf bv = *(const v16bf*)(w2b + (size_t)(c0 + cb) * CCH2 + k0 + khalf * 16);
#pragma unroll
            for (int mt = 0; mt < 4; ++mt) {
                AFrag af;
                const __bf16* pa = &sH[(mt * 16 + cb) * SH_STRIDE + k0 + khalf * 16];
                af.q[0] = *(const u32x4*)pa;
                af.q[1] = *(const u32x4*)(pa + 8);
                acc[mt] = __builtin_amdgcn_wmma_f32_16x16x32_bf16(
                    false, af.v, false, bv, (short)0, acc[mt], false, false);
            }
        }
        float gam = gamma[c0 + cb];
#pragma unroll
        for (int mt = 0; mt < 4; ++mt) {
            int mrow = mt * 16 + khalf * 8;
#pragma unroll
            for (int r = 0; r < 8; ++r) {
                long idx = (long)(row0 + mrow + r) * CCH + c0 + cb;
                out[idx] = X[idx] + acc[mt][r] * gam;
            }
        }
    }
}

// ---------------- host-side launcher ---------------------------------------
extern "C" void kernel_launch(void* const* d_in, const int* in_sizes, int n_in,
                              void* d_out, int out_size, void* d_ws, size_t ws_size,
                              hipStream_t stream) {
    const float* X     = (const float*)d_in[0];
    const float* cw    = (const float*)d_in[1];
    const float* W1    = (const float*)d_in[2];
    const float* W2    = (const float*)d_in[3];
    const float* gamma = (const float*)d_in[4];
    const int*   dil   = (const int*)d_in[5];
    float*       out   = (float*)d_out;

    char*   ws    = (char*)d_ws;
    float*  stats = (float*)ws;                         // 16 f32: (sum, sumsq) per n
    float*  msv   = (float*)(ws + 64);                  // 16 f32: (mean, rstd) per n
    __bf16* w1b   = (__bf16*)(ws + 256);                // 1024x512 bf16
    __bf16* w2b   = (__bf16*)(ws + 256 + (size_t)CCH * CCH2 * 2);  // 512x1024 bf16

    prep_kernel<<<(CCH * CCH2 + 255) / 256, 256, 0, stream>>>(W1, W2, w1b, w2b, stats);
    conv_stats_kernel<<<(MROWS * CCH / 4) / 2048, 256, 0, stream>>>(X, cw, dil, stats);
    finalize_kernel<<<1, 32, 0, stream>>>(stats, msv);

    hipFuncSetAttribute((const void*)mlp_kernel,
                        hipFuncAttributeMaxDynamicSharedMemorySize, LDS_BYTES);
    mlp_kernel<<<MROWS / BM, 256, LDS_BYTES, stream>>>(X, cw, w1b, w2b, gamma, msv, dil, out);
}
